// MultiHeadSelfAttention_24404004176244
// MI455X (gfx1250) — compile-verified, matrix operand order checked by reading, not run on hardware
//
#include <hip/hip_runtime.h>
#include <hip/hip_bf16.h>

// ---------------------------------------------------------------------------
// MultiHeadSelfAttention for MI455X (gfx1250, wave32, WMMA bf16 w/ f32 acc)
//   N_SEQ=2048, BS=8, E=512, H=8, HD=64
// Pipeline:
//   [0] pre-convert seq / W_qkv / W_out to bf16
//   [1] qkv = seq @ Wqkv^T + b   : block = 128(M)x128(N), B strip async-staged
//       to LDS, double-buffered (ASYNCcnt), 8 waves share B via ds_load_b128
//   [2] flash attention per (bs*H) batch; K chunk async global->LDS,
//       V transposed via VGPRs, P round-trips LDS (D-layout -> A-layout)
//   [3] out = attn @ Wout^T + b  : same block-cooperative GEMM, f32 out
// ---------------------------------------------------------------------------

typedef __attribute__((ext_vector_type(16))) __bf16 v16bf;
typedef __attribute__((ext_vector_type(8)))  __bf16 v8bf;
typedef __attribute__((ext_vector_type(8)))  float  v8f;

constexpr int NSEQ = 2048;
constexpr int BS   = 8;
constexpr int E    = 512;
constexpr int H    = 8;
constexpr int HD   = 64;
constexpr int F3   = 3 * E;        // 1536
constexpr int ROWS = NSEQ * BS;    // 16384

__device__ __forceinline__ v8f wmma_bf16(v16bf a, v16bf b, v8f c) {
  return __builtin_amdgcn_wmma_f32_16x16x32_bf16(false, a, false, b, (short)0, c,
                                                 false, false);
}

// A/B 16-bit fragment: lane L (h = L/16) holds k = 8h..8h+7 (elems 0..7) and
// k = 16+8h..16+8h+7 (elems 8..15). Caller passes p already offset by +8h.
__device__ __forceinline__ v16bf load_frag_bf16(const __bf16* p) {
  v8bf lo = *(const v8bf*)(p);
  v8bf hi = *(const v8bf*)(p + 16);
  v16bf f;
#pragma unroll
  for (int i = 0; i < 8; ++i) { f[i] = lo[i]; f[i + 8] = hi[i]; }
  return f;
}

// 32B async global->LDS copy (same IOFFSET applies to both addresses)
__device__ __forceinline__ void async_copy32(unsigned lds_addr, const __bf16* g) {
  asm volatile("global_load_async_to_lds_b128 %0, %1, off"
               :: "v"(lds_addr), "v"(g) : "memory");
  asm volatile("global_load_async_to_lds_b128 %0, %1, off offset:16"
               :: "v"(lds_addr), "v"(g) : "memory");
}

// ---------------------------------------------------------------------------
// Kernel 0: elementwise f32 -> bf16 (8 elems/thread)
// ---------------------------------------------------------------------------
__global__ __launch_bounds__(256) void cvt_bf16(const float* __restrict__ src,
                                                __bf16* __restrict__ dst,
                                                int n8) {
  const int i = blockIdx.x * 256 + threadIdx.x;
  if (i >= n8) return;
  const float4 a = *(const float4*)(src + (size_t)i * 8);
  const float4 b = *(const float4*)(src + (size_t)i * 8 + 4);
  v8bf o;
  o[0] = (__bf16)a.x; o[1] = (__bf16)a.y; o[2] = (__bf16)a.z; o[3] = (__bf16)a.w;
  o[4] = (__bf16)b.x; o[5] = (__bf16)b.y; o[6] = (__bf16)b.z; o[7] = (__bf16)b.w;
  *(v8bf*)(dst + (size_t)i * 8) = o;
}

// ---------------------------------------------------------------------------
// Kernel 1: qkv[r,f] = sum_e seq[r,e]*Wqkv[f,e] + bqkv[f]  (bf16 out)
// Block = 8 waves, each wave one 16-row M tile; all share a 128-col B strip.
// B strip (128 x 32 k, 8KB) async-staged to LDS, double buffered.
// ---------------------------------------------------------------------------
__global__ __launch_bounds__(256) void qkv_gemm(const __bf16* __restrict__ seqb,
                                                const __bf16* __restrict__ Wqkvb,
                                                const float* __restrict__ bqkv,
                                                __bf16* __restrict__ qkv) {
  __shared__ __align__(16) __bf16 blds[2][128 * 32];   // [buf][col][k]
  const int tid  = threadIdx.x;
  const int lane = tid & 31;
  const int wave = tid >> 5;
  const int l16  = lane & 15;
  const int hh   = lane >> 4;
  const int NG   = F3 / 128;              // 12
  const int gn   = blockIdx.x % NG;
  const int gm   = (blockIdx.x / NG) * 8 + wave;
  const int arow = gm * 16 + l16;
  const __bf16* aptr = seqb + (size_t)arow * E + 8 * hh;

  // staging coords: thread t stages 32B of column scol at k-offset shalf
  const int scol  = tid >> 1;             // 0..127
  const int shalf = (tid & 1) * 16;       // 0 or 16 elements
  const __bf16* bsrc = Wqkvb + (size_t)(gn * 128 + scol) * E + shalf;
  const unsigned bl[2] = {
      (unsigned)(uintptr_t)(&blds[0][scol * 32 + shalf]),
      (unsigned)(uintptr_t)(&blds[1][scol * 32 + shalf])};

  v8f acc[8] = {};
  async_copy32(bl[0], bsrc);              // preload k-step 0
  for (int ks = 0; ks < E / 32; ++ks) {
    asm volatile("s_wait_asynccnt 0" ::: "memory");
    __syncthreads();                      // buf[ks&1] ready; prev reads done
    if (ks + 1 < E / 32) async_copy32(bl[(ks + 1) & 1], bsrc + (ks + 1) * 32);
    const __bf16* bb = &blds[ks & 1][0];
    v16bf a = load_frag_bf16(aptr + ks * 32);
#pragma unroll
    for (int j = 0; j < 8; ++j) {
      v16bf b = load_frag_bf16(bb + (j * 16 + l16) * 32 + 8 * hh);
      acc[j] = wmma_bf16(a, b, acc[j]);
    }
  }
#pragma unroll
  for (int j = 0; j < 8; ++j) {
    const int f = gn * 128 + j * 16 + l16;
    const float bias = bqkv[f];
#pragma unroll
    for (int v = 0; v < 8; ++v) {
      const int row = gm * 16 + v + 8 * hh;
      qkv[(size_t)row * F3 + f] = (__bf16)(acc[j][v] + bias);
    }
  }
}

// ---------------------------------------------------------------------------
// Kernel 2: flash attention. Block = 8 waves = 128 q rows of one (bs,head).
// ---------------------------------------------------------------------------
__global__ __launch_bounds__(256) void attn_flash(const __bf16* __restrict__ qkv,
                                                  __bf16* __restrict__ aout) {
  __shared__ __align__(16) __bf16 k_lds[32 * 64];      // [key][d]
  __shared__ __align__(16) __bf16 v_ldsT[64 * 32];     // [d][key]
  __shared__ __align__(16) __bf16 p_lds[8][16 * 32];   // per-wave [row][key]

  const int tid  = threadIdx.x;
  const int lane = tid & 31;
  const int wave = tid >> 5;
  const int l16  = lane & 15;
  const int hh   = lane >> 4;

  const int bh   = blockIdx.x >> 4;       // 0..63  (bs*H + head)
  const int qblk = blockIdx.x & 15;       // 0..15
  const int bsi  = bh >> 3;
  const int head = bh & 7;
  const int qcol = head * HD;
  const int kcol = E + head * HD;
  const int vcol = 2 * E + head * HD;
  const int qbase = qblk * 128 + wave * 16;

  // Q fragments (persistent)
  const int qrow = (qbase + l16) * BS + bsi;
  const v16bf aq0 = load_frag_bf16(qkv + (size_t)qrow * F3 + qcol + 0  + 8 * hh);
  const v16bf aq1 = load_frag_bf16(qkv + (size_t)qrow * F3 + qcol + 32 + 8 * hh);

  v8f o0 = {}, o1 = {}, o2 = {}, o3 = {};
  float mi[8], li[8];
#pragma unroll
  for (int v = 0; v < 8; ++v) { mi[v] = -3.0e38f; li[v] = 0.0f; }

  const int skey = tid >> 3;              // 0..31
  const int sdp  = (tid & 7) << 3;        // 0,8,...,56
  const unsigned klds_addr =
      (unsigned)(uintptr_t)(k_lds + skey * 64 + sdp);

  for (int c = 0; c < NSEQ; c += 32) {
    __syncthreads();  // all waves done reading previous chunk's LDS
    {
      const int grow = (c + skey) * BS + bsi;
      // K chunk: 256 threads x 16B = 4KB, async global -> LDS (ASYNCcnt)
      const __bf16* gk = qkv + (size_t)grow * F3 + kcol + sdp;
      asm volatile("global_load_async_to_lds_b128 %0, %1, off"
                   :: "v"(klds_addr), "v"(gk) : "memory");
      // V chunk: through VGPRs, transposed into v_ldsT[d][key]
      v8bf vv = *(const v8bf*)(qkv + (size_t)grow * F3 + vcol + sdp);
#pragma unroll
      for (int jj = 0; jj < 8; ++jj) v_ldsT[(sdp + jj) * 32 + skey] = vv[jj];
      asm volatile("s_wait_asynccnt 0" ::: "memory");
    }
    __syncthreads();

    // scores: two 16x16 key tiles, K over d (2 steps of 32)
    v8f s0 = {}, s1 = {};
    {
      v16bf b00 = load_frag_bf16(k_lds + (l16) * 64 + 0  + 8 * hh);
      v16bf b01 = load_frag_bf16(k_lds + (l16) * 64 + 32 + 8 * hh);
      v16bf b10 = load_frag_bf16(k_lds + (16 + l16) * 64 + 0  + 8 * hh);
      v16bf b11 = load_frag_bf16(k_lds + (16 + l16) * 64 + 32 + 8 * hh);
      s0 = wmma_bf16(aq0, b00, s0);
      s1 = wmma_bf16(aq0, b10, s1);
      s0 = wmma_bf16(aq1, b01, s0);
      s1 = wmma_bf16(aq1, b11, s1);
    }

    // online softmax, row m = v + 8h lives across the 16-lane half
    __bf16* pw = &p_lds[wave][0];
#pragma unroll
    for (int v = 0; v < 8; ++v) {
      const float x0 = s0[v] * 0.125f;   // 1/sqrt(HD)
      const float x1 = s1[v] * 0.125f;
      float cm = fmaxf(x0, x1);
#pragma unroll
      for (int off = 1; off < 16; off <<= 1)
        cm = fmaxf(cm, __shfl_xor(cm, off, 32));
      const float mnew  = fmaxf(mi[v], cm);
      const float alpha = __expf(mi[v] - mnew);
      const float p0 = __expf(x0 - mnew);
      const float p1 = __expf(x1 - mnew);
      float rs = p0 + p1;
#pragma unroll
      for (int off = 1; off < 16; off <<= 1)
        rs += __shfl_xor(rs, off, 32);
      li[v] = li[v] * alpha + rs;
      mi[v] = mnew;
      o0[v] *= alpha; o1[v] *= alpha; o2[v] *= alpha; o3[v] *= alpha;
      pw[(v + 8 * hh) * 32 + l16]      = (__bf16)p0;
      pw[(v + 8 * hh) * 32 + 16 + l16] = (__bf16)p1;
    }
    asm volatile("s_wait_dscnt 0" ::: "memory");

    // PV: A = P (16 rows x 32 keys), B = V^T column tiles (32 keys x 16 d)
    v16bf pa  = load_frag_bf16(pw + l16 * 32 + 8 * hh);
    v16bf bv0 = load_frag_bf16(v_ldsT + (0  + l16) * 32 + 8 * hh);
    v16bf bv1 = load_frag_bf16(v_ldsT + (16 + l16) * 32 + 8 * hh);
    v16bf bv2 = load_frag_bf16(v_ldsT + (32 + l16) * 32 + 8 * hh);
    v16bf bv3 = load_frag_bf16(v_ldsT + (48 + l16) * 32 + 8 * hh);
    o0 = wmma_bf16(pa, bv0, o0);
    o1 = wmma_bf16(pa, bv1, o1);
    o2 = wmma_bf16(pa, bv2, o2);
    o3 = wmma_bf16(pa, bv3, o3);
  }

  // normalize, store bf16: aout[(n*BS+bs), head*64 + d]
#pragma unroll
  for (int v = 0; v < 8; ++v) {
    const float inv = 1.0f / li[v];
    const int nq   = qbase + v + 8 * hh;
    const int orow = nq * BS + bsi;
    __bf16* dst = aout + (size_t)orow * E + head * HD;
    dst[0 * 16 + l16] = (__bf16)(o0[v] * inv);
    dst[1 * 16 + l16] = (__bf16)(o1[v] * inv);
    dst[2 * 16 + l16] = (__bf16)(o2[v] * inv);
    dst[3 * 16 + l16] = (__bf16)(o3[v] * inv);
  }
}

// ---------------------------------------------------------------------------
// Kernel 3: out[r,f] = sum_e attn[r,e]*Wout[f,e] + bout[f]   (f32 out)
// Same block-cooperative structure as kernel 1.
// ---------------------------------------------------------------------------
__global__ __launch_bounds__(256) void out_gemm(const __bf16* __restrict__ A,
                                                const __bf16* __restrict__ Woutb,
                                                const float* __restrict__ bout,
                                                float* __restrict__ out) {
  __shared__ __align__(16) __bf16 blds[2][128 * 32];
  const int tid  = threadIdx.x;
  const int lane = tid & 31;
  const int wave = tid >> 5;
  const int l16  = lane & 15;
  const int hh   = lane >> 4;
  const int NG   = E / 128;               // 4
  const int gn   = blockIdx.x % NG;
  const int gm   = (blockIdx.x / NG) * 8 + wave;
  const int arow = gm * 16 + l16;
  const __bf16* aptr = A + (size_t)arow * E + 8 * hh;

  const int scol  = tid >> 1;
  const int shalf = (tid & 1) * 16;
  const __bf16* bsrc = Woutb + (size_t)(gn * 128 + scol) * E + shalf;
  const unsigned bl[2] = {
      (unsigned)(uintptr_t)(&blds[0][scol * 32 + shalf]),
      (unsigned)(uintptr_t)(&blds[1][scol * 32 + shalf])};

  v8f acc[8] = {};
  async_copy32(bl[0], bsrc);
  for (int ks = 0; ks < E / 32; ++ks) {
    asm volatile("s_wait_asynccnt 0" ::: "memory");
    __syncthreads();
    if (ks + 1 < E / 32) async_copy32(bl[(ks + 1) & 1], bsrc + (ks + 1) * 32);
    const __bf16* bb = &blds[ks & 1][0];
    v16bf a = load_frag_bf16(aptr + ks * 32);
#pragma unroll
    for (int j = 0; j < 8; ++j) {
      v16bf b = load_frag_bf16(bb + (j * 16 + l16) * 32 + 8 * hh);
      acc[j] = wmma_bf16(a, b, acc[j]);
    }
  }
#pragma unroll
  for (int j = 0; j < 8; ++j) {
    const int f = gn * 128 + j * 16 + l16;
    const float bias = bout[f];
#pragma unroll
    for (int v = 0; v < 8; ++v) {
      const int row = gm * 16 + v + 8 * hh;
      out[(size_t)row * E + f] = acc[j][v] + bias;
    }
  }
}

// ---------------------------------------------------------------------------
extern "C" void kernel_launch(void* const* d_in, const int* in_sizes, int n_in,
                              void* d_out, int out_size, void* d_ws, size_t ws_size,
                              hipStream_t stream) {
  const float* seq  = (const float*)d_in[0];   // [2048, 8, 512]
  const float* Wqkv = (const float*)d_in[1];   // [1536, 512]
  const float* bqkv = (const float*)d_in[2];   // [1536]
  const float* Wout = (const float*)d_in[3];   // [512, 512]
  const float* bout = (const float*)d_in[4];   // [512]
  float* out = (float*)d_out;                  // [2048, 8, 512] f32

  char* ws = (char*)d_ws;
  __bf16* qkv   = (__bf16*)(ws);                          // 50.3 MB
  __bf16* aout  = (__bf16*)(ws + 50331648);               // 16.8 MB
  __bf16* seqb  = (__bf16*)(ws + 67108864);               // 16.8 MB
  __bf16* Wqkvb = (__bf16*)(ws + 83886080);               //  1.5 MB
  __bf16* Woutb = (__bf16*)(ws + 85458944);               //  0.5 MB  (~86 MB)

  // [0] one-shot f32 -> bf16 conversions
  cvt_bf16<<<(ROWS * E / 8) / 256, 256, 0, stream>>>(seq,  seqb,  ROWS * E / 8);
  cvt_bf16<<<(F3 * E / 8) / 256,  256, 0, stream>>>(Wqkv, Wqkvb, F3 * E / 8);
  cvt_bf16<<<(E * E / 8) / 256,   256, 0, stream>>>(Wout, Woutb, E * E / 8);

  // [1] QKV projection: 128 M-blocks x 12 N-groups
  qkv_gemm<<<(ROWS / 128) * (F3 / 128), 256, 0, stream>>>(seqb, Wqkvb, bqkv, qkv);
  // [2] Flash attention: 64 batches x 16 q-blocks (128 rows each)
  attn_flash<<<64 * 16, 256, 0, stream>>>(qkv, aout);
  // [3] Output projection: 128 M-blocks x 4 N-groups
  out_gemm<<<(ROWS / 128) * (E / 128), 256, 0, stream>>>(aout, Woutb, bout, out);
}
